// CapibaraByte_73753178407203
// MI455X (gfx1250) — matrix operand order repaired and checked by reading, not compiled, not run on hardware
//
#include <hip/hip_runtime.h>
#include <hip/hip_bf16.h>

typedef __attribute__((ext_vector_type(16))) __bf16 v16bf;
typedef __attribute__((ext_vector_type(8)))  __bf16 v8bf;
typedef __attribute__((ext_vector_type(8)))  float  v8f;

#define CB_B   128
#define CB_T   1024
#define CB_DIN 256
#define CB_H   2048
#define CB_K   (CB_H + CB_DIN)   // 2304 combined reduction dim
#define NBLK   32                // grid: 32 blocks x 64 cols = 2048
#define NTHR   256               // 8 waves of 32
#define NTILE  4                 // 4 x (16-wide) WMMA tiles per wave
#define UPD    0.1f

static __device__ __forceinline__ float gelu_tanh(float x) {
  const float c = 0.7978845608028654f;  // sqrt(2/pi)
  float x3 = x * x * x;
  return 0.5f * x * (1.0f + tanhf(c * (x + 0.044715f * x3)));
}

static __device__ __forceinline__ v8bf ldv8(const __bf16* p) {
  return *reinterpret_cast<const v8bf*>(p);
}

static __device__ __forceinline__ v16bf cat16(v8bf lo, v8bf hi) {
  return __builtin_shufflevector(lo, hi, 0, 1, 2, 3, 4, 5, 6, 7,
                                 8, 9, 10, 11, 12, 13, 14, 15);
}

// Device-wide barrier: monotonic counter, release on arrive, acquire on spin.
static __device__ __forceinline__ void grid_sync(unsigned* cnt, unsigned target) {
  __threadfence();          // make this thread's stores device-visible
  __syncthreads();
  if (threadIdx.x == 0) {
    __hip_atomic_fetch_add(cnt, 1u, __ATOMIC_ACQ_REL, __HIP_MEMORY_SCOPE_AGENT);
    while (__hip_atomic_load(cnt, __ATOMIC_ACQUIRE, __HIP_MEMORY_SCOPE_AGENT) < target) {
      __builtin_amdgcn_s_sleep(1);
    }
  }
  __syncthreads();
  __threadfence();          // invalidate stale cached lines before next reads
}

// --- prep: Wt[n][k] = bf16( k < H ? W_state[k][n] : W_conv[k-H][n] ) ---
__global__ void capy_prep_w(const float* __restrict__ Ws,
                            const float* __restrict__ Wc,
                            __bf16* __restrict__ wt) {
  int idx = blockIdx.x * blockDim.x + threadIdx.x;
  if (idx >= CB_H * CB_K) return;
  int n = idx / CB_K;
  int k = idx - n * CB_K;
  float v = (k < CB_H) ? Ws[(size_t)k * CB_H + n]
                       : Wc[(size_t)(k - CB_H) * CB_H + n];
  wt[idx] = (__bf16)v;
}

// --- prep: x fp32 -> bf16 (same (B,T,D) layout) ---
__global__ void capy_prep_x(const float* __restrict__ x, __bf16* __restrict__ xb, int n) {
  for (int i = blockIdx.x * blockDim.x + threadIdx.x; i < n; i += gridDim.x * blockDim.x)
    xb[i] = (__bf16)x[i];
}

// --- persistent scan kernel ---
__global__ void __launch_bounds__(NTHR)
capy_scan_kernel(const __bf16* __restrict__ xbf,   // (B,T,DIN) bf16
                 const __bf16* __restrict__ wt,    // (H, K) col-major weights (k contiguous)
                 const float*  __restrict__ bias,  // (H,)
                 float* sF0, float* sF1,           // fp32 state ping-pong (B,H)
                 __bf16* sB0, __bf16* sB1,         // bf16 state ping-pong (B,H)
                 unsigned* counter,
                 float* __restrict__ out) {        // [output(B,H); state(B,H)]
  const int tid  = threadIdx.x;
  const int wave = tid >> 5;
  const int lane = tid & 31;
  const int l16  = lane & 15;
  const bool hi  = lane >= 16;
  const int rowA = wave * 16 + l16;          // A-fragment row (batch index)
  const int n0   = blockIdx.x * (NTILE * 16);
  const int kbA  = hi ? 16 : 0;              // A frag: contiguous K=16h..16h+15 per lane group

  // B-operand column base pointers (lanes 0-15: K=0..15; lanes 16-31: K=16..31)
  const __bf16* bcol[NTILE];
  float bia[NTILE];
#pragma unroll
  for (int nt = 0; nt < NTILE; ++nt) {
    bcol[nt] = wt + (size_t)(n0 + nt * 16 + l16) * CB_K + (hi ? 16 : 0);
    bia[nt]  = bias[n0 + nt * 16 + l16];
  }
  const __bf16* xrow = xbf + (size_t)rowA * CB_T * CB_DIN;

  for (int t = 0; t < CB_T; ++t) {
    const float*  sin_  = (t & 1) ? sF1 : sF0;
    float*        sout  = (t & 1) ? sF0 : sF1;
    const __bf16* sbin  = (t & 1) ? sB1 : sB0;
    __bf16*       sbout = (t & 1) ? sB0 : sB1;

    v8f acc[NTILE];
#pragma unroll
    for (int nt = 0; nt < NTILE; ++nt)
      acc[nt] = (v8f){0.f, 0.f, 0.f, 0.f, 0.f, 0.f, 0.f, 0.f};

    // ---- state @ W_state : K = 0..2047 ----
    const __bf16* arow = sbin + (size_t)rowA * CB_H + kbA;
#pragma unroll 4
    for (int kt = 0; kt < CB_H / 32; ++kt) {
      const int k0 = kt * 32;
      v16bf A = cat16(ldv8(arow + k0), ldv8(arow + k0 + 8));
#pragma unroll
      for (int nt = 0; nt < NTILE; ++nt) {
        v16bf Bf = cat16(ldv8(bcol[nt] + k0), ldv8(bcol[nt] + k0 + 8));
        acc[nt] = __builtin_amdgcn_wmma_f32_16x16x32_bf16(
            false, A, false, Bf, (short)0, acc[nt], false, false);
      }
    }
    // ---- x_t @ W_conv : K = 2048..2303 ----
    const __bf16* axrow = xrow + (size_t)t * CB_DIN + kbA;
#pragma unroll
    for (int kt = 0; kt < CB_DIN / 32; ++kt) {
      const int k0 = kt * 32;
      v16bf A = cat16(ldv8(axrow + k0), ldv8(axrow + k0 + 8));
#pragma unroll
      for (int nt = 0; nt < NTILE; ++nt) {
        v16bf Bf = cat16(ldv8(bcol[nt] + CB_H + k0), ldv8(bcol[nt] + CB_H + k0 + 8));
        acc[nt] = __builtin_amdgcn_wmma_f32_16x16x32_bf16(
            false, A, false, Bf, (short)0, acc[nt], false, false);
      }
    }

    // ---- epilogue: bias + gelu + EMA blend; write fp32 + bf16 state ----
#pragma unroll
    for (int nt = 0; nt < NTILE; ++nt) {
      const int col = n0 + nt * 16 + l16;
#pragma unroll
      for (int r = 0; r < 8; ++r) {
        const int row = wave * 16 + (hi ? 8 : 0) + r;
        const size_t idx = (size_t)row * CB_H + col;
        float u  = acc[nt][r] + bia[nt];
        float ns = (1.0f - UPD) * sin_[idx] + UPD * gelu_tanh(u);
        sout[idx]  = ns;
        sbout[idx] = (__bf16)ns;
      }
    }
    grid_sync(counter, (unsigned)(t + 1) * NBLK);
  }

  // ---- final: out = state @ W_state + bias (state in buffer 0 after 1024 steps) ----
  {
    v8f acc[NTILE];
#pragma unroll
    for (int nt = 0; nt < NTILE; ++nt)
      acc[nt] = (v8f){0.f, 0.f, 0.f, 0.f, 0.f, 0.f, 0.f, 0.f};
    const __bf16* arow = sB0 + (size_t)rowA * CB_H + kbA;
#pragma unroll 4
    for (int kt = 0; kt < CB_H / 32; ++kt) {
      const int k0 = kt * 32;
      v16bf A = cat16(ldv8(arow + k0), ldv8(arow + k0 + 8));
#pragma unroll
      for (int nt = 0; nt < NTILE; ++nt) {
        v16bf Bf = cat16(ldv8(bcol[nt] + k0), ldv8(bcol[nt] + k0 + 8));
        acc[nt] = __builtin_amdgcn_wmma_f32_16x16x32_bf16(
            false, A, false, Bf, (short)0, acc[nt], false, false);
      }
    }
#pragma unroll
    for (int nt = 0; nt < NTILE; ++nt) {
      const int col = n0 + nt * 16 + l16;
#pragma unroll
      for (int r = 0; r < 8; ++r) {
        const int row = wave * 16 + (hi ? 8 : 0) + r;
        out[(size_t)row * CB_H + col] = acc[nt][r] + bia[nt];
      }
    }
  }
  // ---- copy final fp32 state into second half of d_out ----
  for (int i = blockIdx.x * NTHR + tid; i < CB_B * CB_H; i += NBLK * NTHR)
    out[(size_t)CB_B * CB_H + i] = sF0[i];
}

extern "C" void kernel_launch(void* const* d_in, const int* in_sizes, int n_in,
                              void* d_out, int out_size, void* d_ws, size_t ws_size,
                              hipStream_t stream) {
  const float* x    = (const float*)d_in[0];  // (B,T,DIN)
  const float* Ws   = (const float*)d_in[1];  // (H,H)
  const float* Wc   = (const float*)d_in[2];  // (DIN,H)
  const float* bias = (const float*)d_in[3];  // (H,)

  // workspace layout (all 256B aligned)
  char* ws = (char*)d_ws;
  const size_t stateF32Bytes = (size_t)CB_B * CB_H * 4;  // 1 MB each
  const size_t stateBfBytes  = (size_t)CB_B * CB_H * 2;  // 512 KB each
  unsigned* counter = (unsigned*)ws;
  float*  sF0 = (float*)(ws + 256);
  float*  sF1 = (float*)(ws + 256 + stateF32Bytes);
  __bf16* sB0 = (__bf16*)(ws + 256 + 2 * stateF32Bytes);
  __bf16* sB1 = (__bf16*)(ws + 256 + 2 * stateF32Bytes + stateBfBytes);
  __bf16* wt  = (__bf16*)(ws + 256 + 2 * stateF32Bytes + 2 * stateBfBytes);
  __bf16* xb  = wt + (size_t)CB_H * CB_K;

  // zero barrier counter + both state ping-pong buffers (one contiguous region)
  hipMemsetAsync(ws, 0, 256 + 2 * stateF32Bytes + 2 * stateBfBytes, stream);

  const int nW = CB_H * CB_K;
  capy_prep_w<<<(nW + NTHR - 1) / NTHR, NTHR, 0, stream>>>(Ws, Wc, wt);
  capy_prep_x<<<4096, NTHR, 0, stream>>>(x, xb, CB_B * CB_T * CB_DIN);

  capy_scan_kernel<<<NBLK, NTHR, 0, stream>>>(xb, wt, bias, sF0, sF1, sB0, sB1,
                                              counter, (float*)d_out);
}
